// GraphTransformerLayer_63531156242781
// MI455X (gfx1250) — matrix operand order repaired and checked by reading, not compiled, not run on hardware
//
#include <hip/hip_runtime.h>
#include <hip/hip_bf16.h>
#include <stdint.h>

#define NTOK   4096
#define GDIM   256
#define HEADS  8
#define HDIM   32
#define DIMFF  1024
#define ATT_SCALE 0.17677669529663687f   // 1/sqrt(32)

typedef __bf16 bf16;
typedef __attribute__((ext_vector_type(16))) __bf16 v16bf;
typedef __attribute__((ext_vector_type(8)))  __bf16 v8bf;
typedef __attribute__((ext_vector_type(8)))  float  v8f;

union V16 { v16bf v; v8bf h[2]; };
union V8F { v8f v; float f[8]; };

static __device__ inline v8f zero8() { v8f z = {0.f,0.f,0.f,0.f,0.f,0.f,0.f,0.f}; return z; }

static __device__ inline v8f wmma_bf16(v16bf a, v16bf b, v8f c) {
  return __builtin_amdgcn_wmma_f32_16x16x32_bf16(false, a, false, b, (short)0, c, false, false);
}

static __device__ inline float gelu_tanh(float x) {
  float x3 = x * x * x;
  return 0.5f * x * (1.0f + tanhf(0.7978845608028654f * (x + 0.044715f * x3)));
}

// A fragment (16x32 bf16, row-major source, leading dim lda elements).
// lanes 0-15 hold row (lane&15), K {0..15}; lanes 16-31 K {16..31}.
static __device__ inline v16bf load_a_frag(const bf16* A, int lda, int m0, int k0, int lane) {
  int row = m0 + (lane & 15);
  int kb  = k0 + ((lane >> 4) << 4);
  V16 r;
  r.h[0] = *(const v8bf*)(A + (size_t)row * lda + kb);
  r.h[1] = *(const v8bf*)(A + (size_t)row * lda + kb + 8);
  return r.v;
}

// B fragment (32x16 bf16) from Bt stored [N x K] row-major:
// element e = B[K=(lane>>4)*16+e][N=lane&15] = Bt[n0+(lane&15)][k0+(lane>>4)*16+e]
static __device__ inline v16bf load_bt_frag(const bf16* Bt, int ldb, int n0, int k0, int lane) {
  int row = n0 + (lane & 15);
  int kb  = k0 + ((lane >> 4) << 4);
  V16 r;
  r.h[0] = *(const v8bf*)(Bt + (size_t)row * ldb + kb);
  r.h[1] = *(const v8bf*)(Bt + (size_t)row * ldb + kb + 8);
  return r.v;
}

// ---------------------------------------------------------------- mask pack
__global__ void pack_mask_kernel(const int* __restrict__ adj, uint32_t* __restrict__ maskbits) {
  int gid = blockIdx.x * blockDim.x + threadIdx.x;
  int bit = (adj[gid] != 0) ? 1 : 0;
  unsigned long long bal = __ballot(bit);                  // wave32 -> low 32 bits
  if ((threadIdx.x & 31) == 0) maskbits[gid >> 5] = (uint32_t)bal;
}

// --------------------------------------------------- weight transpose->bf16
// in: [K x N] fp32   out: [N x K] bf16
__global__ void transpose_to_bf16_kernel(const float* __restrict__ in, bf16* __restrict__ out,
                                         int K, int N) {
  int gid = blockIdx.x * blockDim.x + threadIdx.x;
  if (gid >= K * N) return;
  int n = gid / K, kk = gid % K;
  out[gid] = (bf16)in[(size_t)kk * N + n];
}

// ---------------------------------------------------------------- layernorm
__global__ void layernorm_bf16_kernel(const float* __restrict__ x, const float* __restrict__ g,
                                      const float* __restrict__ b, bf16* __restrict__ out) {
  __shared__ float s1[GDIM], s2[GDIM];
  int row = blockIdx.x, t = threadIdx.x;
  float v = x[(size_t)row * GDIM + t];
  s1[t] = v; s2[t] = v * v;
  __syncthreads();
  for (int off = GDIM / 2; off > 0; off >>= 1) {
    if (t < off) { s1[t] += s1[t + off]; s2[t] += s2[t + off]; }
    __syncthreads();
  }
  float mu  = s1[0] * (1.0f / GDIM);
  float var = s2[0] * (1.0f / GDIM) - mu * mu;
  float inv = rsqrtf(var + 1e-5f);
  out[(size_t)row * GDIM + t] = (bf16)((v - mu) * inv * g[t] + b[t]);
}

// ------------------------------------------------------------------ qkv GEMM
// 32x64 tile per wave, software-pipelined K loop.
// A: h1 bf16 [NTOK x GDIM]; Bt: wqkvT bf16 [768 x GDIM]
// scatter: q[h][n][d], k[h][n][d], vT[h][d][n]
__global__ void gemm_qkv_kernel(const bf16* __restrict__ A, const bf16* __restrict__ Bt,
                                bf16* __restrict__ qb, bf16* __restrict__ kb,
                                bf16* __restrict__ vT) {
  const int Ncols = 3 * GDIM, K = GDIM;
  int lane = threadIdx.x & 31;
  int wid  = (blockIdx.x * blockDim.x + threadIdx.x) >> 5;
  int tiles_n = Ncols >> 6;
  int tm = (wid / tiles_n) << 5;
  int tn = (wid % tiles_n) << 6;
  v8f acc[8];
  #pragma unroll
  for (int i = 0; i < 8; ++i) acc[i] = zero8();

  v16bf a0 = load_a_frag(A, K, tm,      0, lane);
  v16bf a1 = load_a_frag(A, K, tm + 16, 0, lane);
  v16bf b0 = load_bt_frag(Bt, K, tn,      0, lane);
  v16bf b1 = load_bt_frag(Bt, K, tn + 16, 0, lane);
  v16bf b2 = load_bt_frag(Bt, K, tn + 32, 0, lane);
  v16bf b3 = load_bt_frag(Bt, K, tn + 48, 0, lane);

  for (int k0 = 0; k0 < K; k0 += 32) {
    int kn = (k0 + 32 < K) ? (k0 + 32) : 0;   // last-iter reload is dead but branchless
    v16bf na0 = load_a_frag(A, K, tm,      kn, lane);
    v16bf na1 = load_a_frag(A, K, tm + 16, kn, lane);
    v16bf nb0 = load_bt_frag(Bt, K, tn,      kn, lane);
    v16bf nb1 = load_bt_frag(Bt, K, tn + 16, kn, lane);
    v16bf nb2 = load_bt_frag(Bt, K, tn + 32, kn, lane);
    v16bf nb3 = load_bt_frag(Bt, K, tn + 48, kn, lane);
    acc[0] = wmma_bf16(a0, b0, acc[0]);
    acc[1] = wmma_bf16(a0, b1, acc[1]);
    acc[2] = wmma_bf16(a0, b2, acc[2]);
    acc[3] = wmma_bf16(a0, b3, acc[3]);
    acc[4] = wmma_bf16(a1, b0, acc[4]);
    acc[5] = wmma_bf16(a1, b1, acc[5]);
    acc[6] = wmma_bf16(a1, b2, acc[6]);
    acc[7] = wmma_bf16(a1, b3, acc[7]);
    a0 = na0; a1 = na1; b0 = nb0; b1 = nb1; b2 = nb2; b3 = nb3;
  }

  int half = lane >> 4, l15 = lane & 15;
  #pragma unroll
  for (int i = 0; i < 8; ++i) {
    V8F cc; cc.v = acc[i];
    int nb  = tn + (i & 3) * 16;
    int seg = nb >> 8;             // 0=q 1=k 2=v
    int c   = nb & 255;
    int h   = c >> 5;
    int d   = (c & 31) + l15;
    int mb  = tm + (i >> 2) * 16;
    #pragma unroll
    for (int r = 0; r < 8; ++r) {
      int m = mb + r + half * 8;
      bf16 val = (bf16)cc.f[r];
      if (seg == 0)      qb[((size_t)h * NTOK + m) * HDIM + d] = val;
      else if (seg == 1) kb[((size_t)h * NTOK + m) * HDIM + d] = val;
      else               vT[((size_t)h * HDIM + d) * NTOK + m] = val;
    }
  }
}

// ---------------------------------------------------------------- attention
static __device__ inline void load_kfrags(const bf16* kh, int n0, int lane,
                                          v16bf& b0, v16bf& b1) {
  int half = lane >> 4, l15 = lane & 15, kbase = half * 16;
  V16 t;
  int row = n0 + l15;
  t.h[0] = *(const v8bf*)(kh + (size_t)row * HDIM + kbase);
  t.h[1] = *(const v8bf*)(kh + (size_t)row * HDIM + kbase + 8);
  b0 = t.v;
  row += 16;
  t.h[0] = *(const v8bf*)(kh + (size_t)row * HDIM + kbase);
  t.h[1] = *(const v8bf*)(kh + (size_t)row * HDIM + kbase + 8);
  b1 = t.v;
}

static __device__ inline void load_vfrags(const bf16* vh, int n0, int lane,
                                          v16bf& vb0, v16bf& vb1) {
  int half = lane >> 4, l15 = lane & 15, kbase = n0 + half * 16;
  V16 t;
  t.h[0] = *(const v8bf*)(vh + (size_t)l15 * NTOK + kbase);
  t.h[1] = *(const v8bf*)(vh + (size_t)l15 * NTOK + kbase + 8);
  vb0 = t.v;
  t.h[0] = *(const v8bf*)(vh + (size_t)(16 + l15) * NTOK + kbase);
  t.h[1] = *(const v8bf*)(vh + (size_t)(16 + l15) * NTOK + kbase + 8);
  vb1 = t.v;
}

// one wave per (head, 16-row tile); flash-style online softmax over 32-key chunks
__global__ void attn_kernel(const bf16* __restrict__ q, const bf16* __restrict__ k,
                            const bf16* __restrict__ vT, const uint32_t* __restrict__ maskbits,
                            bf16* __restrict__ attn_out) {
  __shared__ __align__(16) bf16 plds[8][16 * 32];
  int lane   = threadIdx.x & 31;
  int wlocal = threadIdx.x >> 5;
  int wid    = (blockIdx.x * blockDim.x + threadIdx.x) >> 5;   // 0..2047
  int h  = wid >> 8;
  int m0 = (wid & 255) << 4;
  const bf16* qh = q  + (size_t)h * NTOK * HDIM;
  const bf16* kh = k  + (size_t)h * NTOK * HDIM;
  const bf16* vh = vT + (size_t)h * HDIM * NTOK;
  int half = lane >> 4, l15 = lane & 15;
  bf16* pl = plds[wlocal];

  V16 qa;
  {
    int row = m0 + l15, kbase = half * 16;
    qa.h[0] = *(const v8bf*)(qh + (size_t)row * HDIM + kbase);
    qa.h[1] = *(const v8bf*)(qh + (size_t)row * HDIM + kbase + 8);
  }

  V8F A0, A1; A0.v = zero8(); A1.v = zero8();
  float rm[8], rs[8];
  #pragma unroll
  for (int r = 0; r < 8; ++r) { rm[r] = -1e30f; rs[r] = 0.0f; }

  v16bf b0, b1;
  load_kfrags(kh, 0, lane, b0, b1);               // pipeline preload

  for (int n0 = 0; n0 < NTOK; n0 += 32) {
    V8F s0, s1;
    s0.v = wmma_bf16(qa.v, b0, zero8());
    s1.v = wmma_bf16(qa.v, b1, zero8());

    // issue next-chunk K loads and this-chunk V loads before the softmax VALU block
    int nn = (n0 + 32 < NTOK) ? (n0 + 32) : 0;
    v16bf nb0, nb1, vb0, vb1;
    load_kfrags(kh, nn, lane, nb0, nb1);
    load_vfrags(vh, n0, lane, vb0, vb1);

    int wcol = n0 >> 5;
    V8F p0, p1;
    #pragma unroll
    for (int r = 0; r < 8; ++r) {
      int m = m0 + r + half * 8;
      uint32_t wm = maskbits[(size_t)m * (NTOK / 32) + wcol];
      float sa = ((wm >> l15) & 1u)        ? s0.f[r] * ATT_SCALE : -1e30f;
      float sb = ((wm >> (l15 + 16)) & 1u) ? s1.f[r] * ATT_SCALE : -1e30f;
      float cm = fmaxf(sa, sb);
      cm = fmaxf(cm, __shfl_xor(cm, 8, 32));
      cm = fmaxf(cm, __shfl_xor(cm, 4, 32));
      cm = fmaxf(cm, __shfl_xor(cm, 2, 32));
      cm = fmaxf(cm, __shfl_xor(cm, 1, 32));
      float mnew = fmaxf(rm[r], cm);
      float corr = __expf(rm[r] - mnew);
      rm[r] = mnew;
      float e0 = __expf(sa - mnew);
      float e1 = __expf(sb - mnew);
      float ls = e0 + e1;
      ls += __shfl_xor(ls, 8, 32);
      ls += __shfl_xor(ls, 4, 32);
      ls += __shfl_xor(ls, 2, 32);
      ls += __shfl_xor(ls, 1, 32);
      rs[r] = rs[r] * corr + ls;
      A0.f[r] *= corr;
      A1.f[r] *= corr;
      p0.f[r] = e0;
      p1.f[r] = e1;
    }
    // C-layout -> A-layout transpose of P through per-wave LDS
    #pragma unroll
    for (int r = 0; r < 8; ++r) {
      int prow = r + half * 8;
      pl[prow * 32 + l15]      = (bf16)p0.f[r];
      pl[prow * 32 + 16 + l15] = (bf16)p1.f[r];
    }
    V16 pa;
    pa.h[0] = *(const v8bf*)(pl + l15 * 32 + half * 16);
    pa.h[1] = *(const v8bf*)(pl + l15 * 32 + half * 16 + 8);

    A0.v = wmma_bf16(pa.v, vb0, A0.v);
    A1.v = wmma_bf16(pa.v, vb1, A1.v);
    b0 = nb0; b1 = nb1;
  }

  #pragma unroll
  for (int r = 0; r < 8; ++r) {
    int m = m0 + r + half * 8;
    float inv = 1.0f / rs[r];
    attn_out[(size_t)m * GDIM + h * HDIM + l15]      = (bf16)(A0.f[r] * inv);
    attn_out[(size_t)m * GDIM + h * HDIM + 16 + l15] = (bf16)(A1.f[r] * inv);
  }
}

// ---------------------------------------------------------- generic GEMM
// 32x64 tile per wave, software-pipelined K loop.
// out = epilogue(A[MxK]bf16 @ Bt[NxK]bf16^T + bias) (+resid); fp32 and/or bf16 out
__global__ void gemm_epi_kernel(const bf16* __restrict__ A, const bf16* __restrict__ Bt,
                                const float* __restrict__ bias, const float* __restrict__ resid,
                                float* __restrict__ outf, bf16* __restrict__ outb,
                                int M, int N, int K, int gelu_flag) {
  int lane = threadIdx.x & 31;
  int wid  = (blockIdx.x * blockDim.x + threadIdx.x) >> 5;
  int tiles_n = N >> 6;
  int tm = (wid / tiles_n) << 5;
  int tn = (wid % tiles_n) << 6;
  if (tm >= M) return;
  v8f acc[8];
  #pragma unroll
  for (int i = 0; i < 8; ++i) acc[i] = zero8();

  v16bf a0 = load_a_frag(A, K, tm,      0, lane);
  v16bf a1 = load_a_frag(A, K, tm + 16, 0, lane);
  v16bf b0 = load_bt_frag(Bt, K, tn,      0, lane);
  v16bf b1 = load_bt_frag(Bt, K, tn + 16, 0, lane);
  v16bf b2 = load_bt_frag(Bt, K, tn + 32, 0, lane);
  v16bf b3 = load_bt_frag(Bt, K, tn + 48, 0, lane);

  for (int k0 = 0; k0 < K; k0 += 32) {
    int kn = (k0 + 32 < K) ? (k0 + 32) : 0;
    v16bf na0 = load_a_frag(A, K, tm,      kn, lane);
    v16bf na1 = load_a_frag(A, K, tm + 16, kn, lane);
    v16bf nb0 = load_bt_frag(Bt, K, tn,      kn, lane);
    v16bf nb1 = load_bt_frag(Bt, K, tn + 16, kn, lane);
    v16bf nb2 = load_bt_frag(Bt, K, tn + 32, kn, lane);
    v16bf nb3 = load_bt_frag(Bt, K, tn + 48, kn, lane);
    acc[0] = wmma_bf16(a0, b0, acc[0]);
    acc[1] = wmma_bf16(a0, b1, acc[1]);
    acc[2] = wmma_bf16(a0, b2, acc[2]);
    acc[3] = wmma_bf16(a0, b3, acc[3]);
    acc[4] = wmma_bf16(a1, b0, acc[4]);
    acc[5] = wmma_bf16(a1, b1, acc[5]);
    acc[6] = wmma_bf16(a1, b2, acc[6]);
    acc[7] = wmma_bf16(a1, b3, acc[7]);
    a0 = na0; a1 = na1; b0 = nb0; b1 = nb1; b2 = nb2; b3 = nb3;
  }

  int half = lane >> 4, l15 = lane & 15;
  #pragma unroll
  for (int i = 0; i < 8; ++i) {
    V8F cc; cc.v = acc[i];
    int n  = tn + (i & 3) * 16 + l15;
    int mb = tm + (i >> 2) * 16;
    float bv = bias ? bias[n] : 0.0f;
    #pragma unroll
    for (int r = 0; r < 8; ++r) {
      int m = mb + r + half * 8;
      float v = cc.f[r] + bv;
      if (gelu_flag) v = gelu_tanh(v);
      if (resid) v += resid[(size_t)m * N + n];
      if (outf) outf[(size_t)m * N + n] = v;
      if (outb) outb[(size_t)m * N + n] = (bf16)v;
    }
  }
}

// ------------------------------------------------------------------- launch
extern "C" void kernel_launch(void* const* d_in, const int* in_sizes, int n_in,
                              void* d_out, int out_size, void* d_ws, size_t ws_size,
                              hipStream_t stream) {
  const float* x      = (const float*)d_in[0];
  const int*   adj    = (const int*)d_in[1];
  const float* w_qkv  = (const float*)d_in[2];
  const float* w_proj = (const float*)d_in[3];
  const float* b_proj = (const float*)d_in[4];
  const float* w1     = (const float*)d_in[5];
  const float* b1     = (const float*)d_in[6];
  const float* w2     = (const float*)d_in[7];
  const float* b2     = (const float*)d_in[8];
  const float* g1     = (const float*)d_in[9];
  const float* be1    = (const float*)d_in[10];
  const float* g2     = (const float*)d_in[11];
  const float* be2    = (const float*)d_in[12];
  float* out = (float*)d_out;

  char* wsp = (char*)d_ws;
  auto carve = [&](size_t bytes) -> char* {
    char* p = wsp; wsp += (bytes + 255) & ~(size_t)255; return p;
  };
  uint32_t* maskbits = (uint32_t*)carve((size_t)NTOK * NTOK / 8);        // 2 MB
  bf16* h1       = (bf16*)carve((size_t)NTOK * GDIM * 2);               // 2 MB
  bf16* h2       = (bf16*)carve((size_t)NTOK * GDIM * 2);               // 2 MB
  bf16* wqkvT    = (bf16*)carve((size_t)3 * GDIM * GDIM * 2);
  bf16* wprojT   = (bf16*)carve((size_t)GDIM * GDIM * 2);
  bf16* w1T      = (bf16*)carve((size_t)DIMFF * GDIM * 2);
  bf16* w2T      = (bf16*)carve((size_t)GDIM * DIMFF * 2);
  bf16* qb       = (bf16*)carve((size_t)NTOK * GDIM * 2);
  bf16* kb       = (bf16*)carve((size_t)NTOK * GDIM * 2);
  bf16* vTb      = (bf16*)carve((size_t)NTOK * GDIM * 2);
  bf16* attn_out = (bf16*)carve((size_t)NTOK * GDIM * 2);
  float* x1      = (float*)carve((size_t)NTOK * GDIM * 4);              // 4 MB
  bf16* a1       = (bf16*)carve((size_t)NTOK * DIMFF * 2);              // 8 MB

  // 1) adj -> bitmask (read 64 MB exactly once)
  pack_mask_kernel<<<(NTOK * NTOK) / 256, 256, 0, stream>>>(adj, maskbits);
  // 2) weights -> bf16 transposed [N x K]
  transpose_to_bf16_kernel<<<(GDIM * 3 * GDIM + 255) / 256, 256, 0, stream>>>(w_qkv,  wqkvT,  GDIM,  3 * GDIM);
  transpose_to_bf16_kernel<<<(GDIM * GDIM + 255) / 256, 256, 0, stream>>>(w_proj, wprojT, GDIM,  GDIM);
  transpose_to_bf16_kernel<<<(GDIM * DIMFF + 255) / 256, 256, 0, stream>>>(w1,     w1T,    GDIM,  DIMFF);
  transpose_to_bf16_kernel<<<(DIMFF * GDIM + 255) / 256, 256, 0, stream>>>(w2,     w2T,    DIMFF, GDIM);
  // 3) LN1
  layernorm_bf16_kernel<<<NTOK, GDIM, 0, stream>>>(x, g1, be1, h1);
  // 4) qkv GEMM + head scatter: waves = (4096/32)*(768/64) = 1536
  gemm_qkv_kernel<<<1536 / 8, 256, 0, stream>>>(h1, wqkvT, qb, kb, vTb);
  // 5) attention (2048 waves)
  attn_kernel<<<2048 / 8, 256, 0, stream>>>(qb, kb, vTb, maskbits, attn_out);
  // 6) proj + residual -> x1 (fp32): waves = 128*4 = 512
  gemm_epi_kernel<<<512 / 8, 256, 0, stream>>>(attn_out, wprojT, b_proj, x, x1, nullptr,
                                               NTOK, GDIM, GDIM, 0);
  // 7) LN2
  layernorm_bf16_kernel<<<NTOK, GDIM, 0, stream>>>(x1, g2, be2, h2);
  // 8) FF1 + GELU -> a1 (bf16): waves = 128*16 = 2048
  gemm_epi_kernel<<<2048 / 8, 256, 0, stream>>>(h2, w1T, b1, nullptr, nullptr, a1,
                                                NTOK, DIMFF, GDIM, 1);
  // 9) FF2 + residual -> out (fp32): waves = 128*4 = 512
  gemm_epi_kernel<<<512 / 8, 256, 0, stream>>>(a1, w2T, b2, x1, out, nullptr,
                                               NTOK, GDIM, DIMFF, 0);
}
